// ConceptualDeltaNetLayer_11244224381578
// MI455X (gfx1250) — compile-verified, matrix operand order checked by reading, not run on hardware
//
#include <hip/hip_runtime.h>
#include <hip/hip_bf16.h>
#include <cstdint>
#include <cstddef>

typedef __bf16 bf16;
typedef __attribute__((ext_vector_type(16))) __bf16 v16bf;
typedef __attribute__((ext_vector_type(8)))  float  v8f;
typedef __attribute__((ext_vector_type(4)))  int    v4i;

#define Bn 4
#define Ln 2048
#define Dn 1024
#define Hn 8
#define HDn 128
#define EPSf 1e-6f

// ---------- async global->LDS (CDNA5) with portable fallback ----------

#if __has_builtin(__builtin_amdgcn_global_load_async_to_lds_b128) && \
    __has_builtin(__builtin_amdgcn_s_wait_asynccnt)
#define USE_ASYNC_LDS 1
#else
#define USE_ASYNC_LDS 0
#endif

#if USE_ASYNC_LDS
typedef __attribute__((address_space(1))) v4i gv4i;
typedef __attribute__((address_space(3))) v4i lv4i;
#endif

__device__ inline void async_cp16(const void* g, void* l) {
#if USE_ASYNC_LDS
  __builtin_amdgcn_global_load_async_to_lds_b128((gv4i*)g, (lv4i*)l, 0, 0);
#else
  *(uint4*)l = *(const uint4*)g;
#endif
}

__device__ inline void wait_async_le8() {
#if USE_ASYNC_LDS
  __builtin_amdgcn_s_wait_asynccnt(8);
#endif
}

__device__ inline void wait_async_le0() {
#if USE_ASYNC_LDS
  __builtin_amdgcn_s_wait_asynccnt(0);
#endif
}

// ---------- helpers ----------

__device__ inline v8f vzero8() {
  v8f z;
#pragma unroll
  for (int e = 0; e < 8; ++e) z[e] = 0.f;
  return z;
}

__device__ inline v8f wmma_bf(v16bf a, v16bf b, v8f c) {
  // D = A(16x32 bf16) * B(32x16 bf16) + C(16x16 f32)
  return __builtin_amdgcn_wmma_f32_16x16x32_bf16(false, a, false, b, (short)0, c,
                                                 false, false);
}

// Gather a 16x32 bf16 operand fragment from a row-major [16+][ld] bf16 array.
// Works for A operand (rows) and B operand (when array holds B^T rows).
__device__ inline v16bf frag_ld(const bf16* p0, int ld) {
  int lane = threadIdx.x & 31;
  int r  = lane & 15;
  int k0 = (lane & 16) ? 8 : 0;
  const bf16* p = p0 + (size_t)r * ld + k0;
  v16bf f;
#pragma unroll
  for (int i = 0; i < 8; ++i) f[i] = p[i];
#pragma unroll
  for (int i = 0; i < 8; ++i) f[8 + i] = p[16 + i];
  return f;
}

__device__ inline float wave_sum(float x) {
#pragma unroll
  for (int off = 16; off > 0; off >>= 1) x += __shfl_xor(x, off, 32);
  return x;
}

// ---------- fp32 -> bf16 pack ----------

__global__ __launch_bounds__(256) void pack_bf16(const float* __restrict__ s,
                                                 bf16* __restrict__ d) {
  const size_t i = ((size_t)blockIdx.x * 256 + threadIdx.x) * 4;
  float4 v = *(const float4*)(s + i);
  d[i + 0] = (bf16)v.x; d[i + 1] = (bf16)v.y;
  d[i + 2] = (bf16)v.z; d[i + 3] = (bf16)v.w;
}

// ---------- GEMM: C[M,N] = A[M,K] * W[N,K]^T + bias (A,W bf16; async-staged) ----------
// 128x128 block tile, K-step 64, double-buffered async global->LDS.

__global__ __launch_bounds__(256) void gemm_bf(const bf16* __restrict__ A,
                                               const bf16* __restrict__ W,
                                               const float* __restrict__ bias,
                                               float* __restrict__ C,
                                               int M, int N, int K) {
  extern __shared__ char gsm[];
  bf16* As = (bf16*)gsm;                // [2][128*64]
  bf16* Ws = (bf16*)(gsm + 32768);      // [2][128*64]
  const int tid = threadIdx.x;
  const int lane = tid & 31, wave = tid >> 5;
  const int wm = wave & 3, wn = wave >> 2;  // 4x2 wave grid -> 32x64 per wave
  const int row0 = blockIdx.y * 128;
  const int col0 = blockIdx.x * 128;

  v8f acc[2][4];
#pragma unroll
  for (int i = 0; i < 2; ++i)
#pragma unroll
    for (int j = 0; j < 4; ++j) acc[i][j] = vzero8();

  const int S = K / 64;
  // prologue: stage 0
  for (int i = tid; i < 1024; i += 256) {
    int r = i >> 3, c8 = (i & 7) << 3;  // 16B chunks of a [128][64] bf16 tile
    async_cp16(A + (size_t)(row0 + r) * K + c8, As + r * 64 + c8);
    async_cp16(W + (size_t)(col0 + r) * K + c8, Ws + r * 64 + c8);
  }

  for (int s = 0; s < S; ++s) {
    if (s + 1 < S) {
      bf16* dA = As + ((s + 1) & 1) * (128 * 64);
      bf16* dW = Ws + ((s + 1) & 1) * (128 * 64);
      const int kt = (s + 1) * 64;
      for (int i = tid; i < 1024; i += 256) {
        int r = i >> 3, c8 = (i & 7) << 3;
        async_cp16(A + (size_t)(row0 + r) * K + kt + c8, dA + r * 64 + c8);
        async_cp16(W + (size_t)(col0 + r) * K + kt + c8, dW + r * 64 + c8);
      }
      wait_async_le8();   // in-order: stage s complete
    } else {
      wait_async_le0();
    }
    __syncthreads();

    const bf16* cA = As + (s & 1) * (128 * 64);
    const bf16* cW = Ws + (s & 1) * (128 * 64);
#pragma unroll
    for (int ks = 0; ks < 2; ++ks) {
      v16bf a0 = frag_ld(cA + (wm * 32 + 0) * 64 + ks * 32, 64);
      v16bf a1 = frag_ld(cA + (wm * 32 + 16) * 64 + ks * 32, 64);
#pragma unroll
      for (int j = 0; j < 4; ++j) {
        v16bf b = frag_ld(cW + (wn * 64 + j * 16) * 64 + ks * 32, 64);
        acc[0][j] = wmma_bf(a0, b, acc[0][j]);
        acc[1][j] = wmma_bf(a1, b, acc[1][j]);
      }
    }
    __syncthreads();
  }

#pragma unroll
  for (int i = 0; i < 2; ++i)
#pragma unroll
    for (int j = 0; j < 4; ++j)
#pragma unroll
      for (int e = 0; e < 8; ++e) {
        int r = (lane < 16) ? e : e + 8;
        int c = lane & 15;
        int gr = row0 + wm * 32 + i * 16 + r;
        int gc = col0 + wn * 64 + j * 16 + c;
        C[(size_t)gr * N + gc] = acc[i][j][e] + bias[gc];
      }
}

// ---------- beta = sigmoid(x @ Wbeta^T + bbeta) ----------

__global__ __launch_bounds__(256) void beta_proj(const float* __restrict__ x,
                                                 const float* __restrict__ Wb,
                                                 const float* __restrict__ bb,
                                                 float* __restrict__ beta_raw) {
  __shared__ float Wsm[Hn * Dn];
  const int tid = threadIdx.x;
  for (int i = tid; i < Hn * Dn; i += 256) Wsm[i] = Wb[i];
  __syncthreads();
  const int row = tid >> 3, h = tid & 7;
  const size_t grow = (size_t)blockIdx.x * 32 + row;
  const float* xr = x + grow * Dn;
  const float* wr = Wsm + h * Dn;
  float acc = bb[h];
  for (int i = 0; i < Dn; i += 4) {
    float4 xv = *(const float4*)(xr + i);
    float4 wv = *(const float4*)(wr + i);
    acc += xv.x * wv.x + xv.y * wv.y + xv.z * wv.z + xv.w * wv.w;
  }
  beta_raw[grow * Hn + h] = 1.f / (1.f + __expf(-acc));
}

// ---------- causal depthwise conv (4 taps) ----------

__global__ __launch_bounds__(256) void dwconv(const float* __restrict__ in,
                                              const float* __restrict__ w,
                                              const float* __restrict__ b,
                                              float* __restrict__ out) {
  const size_t gid = (size_t)blockIdx.x * 256 + threadIdx.x;
  const int d = (int)(gid % Dn);
  const int l = (int)((gid / Dn) % Ln);
  const int bi = (int)(gid / ((size_t)Dn * Ln));
  float acc = b[d];
  const float4 wv = *(const float4*)(w + d * 4);
  const float tw[4] = {wv.x, wv.y, wv.z, wv.w};
  const float* base = in + (size_t)bi * Ln * Dn + d;
#pragma unroll
  for (int j = 0; j < 4; ++j) {
    int ls = l - 3 + j;
    if (ls >= 0) acc += tw[j] * base[(size_t)ls * Dn];
  }
  out[gid] = acc;
}

// ---------- per-(b,l,h): silu, norms, beta scaling; emit [BH,L,HD] ----------

__global__ __launch_bounds__(256) void prep(const float* __restrict__ qc,
                                            const float* __restrict__ kc,
                                            const float* __restrict__ vc,
                                            const float* __restrict__ beta_raw,
                                            bf16* __restrict__ q_r,
                                            bf16* __restrict__ k_r,
                                            float* __restrict__ v_r,
                                            float* __restrict__ beta_s) {
  const int tid = threadIdx.x, lane = tid & 31;
  const int gw = blockIdx.x * 8 + (tid >> 5);
  const int l = gw % Ln;
  const int bh = gw / Ln;
  const int b = bh >> 3, h = bh & 7;

  const size_t src = ((size_t)(b * Ln + l)) * Dn + h * HDn + lane * 4;
  float4 qv = *(const float4*)(qc + src);
  float4 kv = *(const float4*)(kc + src);
  float4 vv = *(const float4*)(vc + src);

  vv.x = vv.x / (1.f + __expf(-vv.x));
  vv.y = vv.y / (1.f + __expf(-vv.y));
  vv.z = vv.z / (1.f + __expf(-vv.z));
  vv.w = vv.w / (1.f + __expf(-vv.w));

  float ks2 = kv.x * kv.x + kv.y * kv.y + kv.z * kv.z + kv.w * kv.w;
  float qs2 = qv.x * qv.x + qv.y * qv.y + qv.z * qv.z + qv.w * qv.w;
  float kn = sqrtf(wave_sum(ks2));
  float qn = sqrtf(wave_sum(qs2));
  float kinv = 1.f / (kn + EPSf);
  float qinv = 1.f / (qn + EPSf);

  const size_t dst = ((size_t)bh * Ln + l) * HDn + lane * 4;
  q_r[dst + 0] = (bf16)(qv.x * qinv); q_r[dst + 1] = (bf16)(qv.y * qinv);
  q_r[dst + 2] = (bf16)(qv.z * qinv); q_r[dst + 3] = (bf16)(qv.w * qinv);
  k_r[dst + 0] = (bf16)(kv.x * kinv); k_r[dst + 1] = (bf16)(kv.y * kinv);
  k_r[dst + 2] = (bf16)(kv.z * kinv); k_r[dst + 3] = (bf16)(kv.w * kinv);
  float4 vo;
  vo.x = vv.x * kinv; vo.y = vv.y * kinv; vo.z = vv.z * kinv; vo.w = vv.w * kinv;
  *(float4*)(v_r + dst) = vo;

  if (lane == 0) {
    float knr = kn * kinv;  // ||k|| after normalization
    float kn2 = knr * knr;
    float br = beta_raw[((size_t)(b * Ln + l)) * Hn + h];
    beta_s[(size_t)bh * Ln + l] = br / (1.f + br * kn2);  // ETA=1
  }
}

// ---------- chunked delta-rule scan, one block per (b,h), chunk C=32 ----------
// S (128x128 fp32) lives in WMMA accumulators; wave w owns S columns w*16..w*16+15.

#define SMEM_BYTES 137344

__global__ __launch_bounds__(256) void delta_scan(const bf16* __restrict__ qg,
                                                  const bf16* __restrict__ kg,
                                                  const float* __restrict__ vg,
                                                  const float* __restrict__ betag,
                                                  float* __restrict__ og) {
  extern __shared__ char smem[];
  bf16*  S_t  = (bf16*)(smem);            // [128][128]  S^T (S_t[n][m]=S[m][n])
  bf16*  Qc   = (bf16*)(smem + 32768);    // [32][128]
  bf16*  Kc   = (bf16*)(smem + 40960);    // [32][128]
  bf16*  Kt   = (bf16*)(smem + 49152);    // [128][32]   K^T
  bf16*  Ut   = (bf16*)(smem + 57344);    // [128][32]   U^T
  bf16*  Mm   = (bf16*)(smem + 65536);    // [32][32]    tril-incl(Q K^T)
  float* Vc   = (float*)(smem + 67584);   // [32][128]
  float* Uf   = (float*)(smem + 83968);   // [32][128]
  float* QS0  = (float*)(smem + 100352);  // [32][128]
  float* KS0  = (float*)(smem + 116736);  // [32][128]
  float* Af   = (float*)(smem + 133120);  // [32][32]    K K^T
  float* betc = (float*)(smem + 137216);  // [32]

  const int tid = threadIdx.x, lane = tid & 31, wave = tid >> 5;
  const int bh = blockIdx.x;

  v8f sacc[8];
#pragma unroll
  for (int fr = 0; fr < 8; ++fr) sacc[fr] = vzero8();

  for (int ch = 0; ch < Ln / 32; ++ch) {
    const int t0 = ch * 32;

    // P0: dump S (registers, fp32) -> S_t (LDS, bf16, transposed)
#pragma unroll
    for (int fr = 0; fr < 8; ++fr)
#pragma unroll
      for (int e = 0; e < 8; ++e) {
        int r = (lane < 16) ? e : e + 8;
        int c = lane & 15;
        S_t[(wave * 16 + c) * 128 + (fr * 16 + r)] = (bf16)sacc[fr][e];
      }

    // P1: stage chunk via async global->LDS copies
    {
      const size_t base = ((size_t)bh * Ln + t0) * HDn;
      for (int i = tid; i < 512; i += 256) {            // 8 bf16 per 16B chunk
        async_cp16(qg + base + i * 8, Qc + i * 8);
        async_cp16(kg + base + i * 8, Kc + i * 8);
      }
      for (int i = tid; i < 1024; i += 256)             // 4 fp32 per 16B chunk
        async_cp16(vg + base + i * 4, Vc + i * 4);
      if (tid < 32) betc[tid] = betag[(size_t)bh * Ln + t0 + tid];
      wait_async_le0();
    }
    __syncthreads();
    // build K^T from Kc
    for (int i = tid; i < 32 * 128; i += 256) {
      int r = i >> 7, c = i & 127;
      Kt[c * 32 + r] = Kc[i];
    }
    __syncthreads();

    // P2a: QS0 = Q*S, KS0 = K*S  (B operand from S_t rows)
    {
      v8f aq[2], ak[2];
      aq[0] = vzero8(); aq[1] = vzero8(); ak[0] = vzero8(); ak[1] = vzero8();
#pragma unroll
      for (int ks = 0; ks < 4; ++ks) {
        v16bf bS = frag_ld(S_t + (wave * 16) * 128 + ks * 32, 128);
#pragma unroll
        for (int fr = 0; fr < 2; ++fr) {
          v16bf aQ = frag_ld(Qc + (fr * 16) * 128 + ks * 32, 128);
          aq[fr] = wmma_bf(aQ, bS, aq[fr]);
          v16bf aK = frag_ld(Kc + (fr * 16) * 128 + ks * 32, 128);
          ak[fr] = wmma_bf(aK, bS, ak[fr]);
        }
      }
#pragma unroll
      for (int fr = 0; fr < 2; ++fr)
#pragma unroll
        for (int e = 0; e < 8; ++e) {
          int r = (lane < 16) ? e : e + 8;
          int c = lane & 15;
          QS0[(fr * 16 + r) * 128 + wave * 16 + c] = aq[fr][e];
          KS0[(fr * 16 + r) * 128 + wave * 16 + c] = ak[fr][e];
        }
    }
    // P2b: Af = K K^T (waves 0-3);  Mm = tril(Q K^T) (waves 4-7)
    {
      const int fi = (wave & 3) >> 1, fj = wave & 1;
      const bf16* Asrc = (wave < 4) ? Kc : Qc;
      v8f am = vzero8();
#pragma unroll
      for (int ks = 0; ks < 4; ++ks) {
        v16bf aA = frag_ld(Asrc + (fi * 16) * 128 + ks * 32, 128);
        v16bf bB = frag_ld(Kc + (fj * 16) * 128 + ks * 32, 128);
        am = wmma_bf(aA, bB, am);
      }
#pragma unroll
      for (int e = 0; e < 8; ++e) {
        int r = (lane < 16) ? e : e + 8;
        int c = lane & 15;
        int t = fi * 16 + r, s = fj * 16 + c;
        if (wave < 4) Af[t * 32 + s] = am[e];
        else          Mm[t * 32 + s] = (s <= t) ? (bf16)am[e] : (bf16)0.f;
      }
    }
    __syncthreads();

    // P3: forward substitution  u_t = beta_t*(v_t - KS0_t - sum_{s<t} Af[t,s] u_s)
    for (int t = 0; t < 32; ++t) {
      if (tid < 128) {
        float acc = Vc[t * 128 + tid] - KS0[t * 128 + tid];
        for (int s = 0; s < t; ++s) acc -= Af[t * 32 + s] * Uf[s * 128 + tid];
        float u = betc[t] * acc;
        Uf[t * 128 + tid] = u;
        Ut[tid * 32 + t] = (bf16)u;
      }
      __syncthreads();
    }

    // P4: O = QS0 + tril(M)*U ; P5: S += K^T * U
    {
      v16bf bU = frag_ld(Ut + (wave * 16) * 32, 32);
#pragma unroll
      for (int fr = 0; fr < 2; ++fr) {
        v8f o;
#pragma unroll
        for (int e = 0; e < 8; ++e) {
          int r = (lane < 16) ? e : e + 8;
          int c = lane & 15;
          o[e] = QS0[(fr * 16 + r) * 128 + wave * 16 + c];
        }
        v16bf aM = frag_ld(Mm + (fr * 16) * 32, 32);
        o = wmma_bf(aM, bU, o);
#pragma unroll
        for (int e = 0; e < 8; ++e) {
          int r = (lane < 16) ? e : e + 8;
          int c = lane & 15;
          og[((size_t)bh * Ln + t0 + fr * 16 + r) * HDn + wave * 16 + c] = o[e];
        }
      }
#pragma unroll
      for (int fr = 0; fr < 8; ++fr) {
        v16bf aK = frag_ld(Kt + (fr * 16) * 32, 32);
        sacc[fr] = wmma_bf(aK, bU, sacc[fr]);
      }
    }
    __syncthreads();
  }
}

// ---------- RMSNorm over head_dim, rearrange [BH,L,HD] -> [B,L,D] (bf16 out) ----------

__global__ __launch_bounds__(256) void rmsnorm_k(const float* __restrict__ og,
                                                 const float* __restrict__ w,
                                                 bf16* __restrict__ on) {
  const int tid = threadIdx.x, lane = tid & 31;
  const int gw = blockIdx.x * 8 + (tid >> 5);
  const int l = gw % Ln, bh = gw / Ln;
  const int b = bh >> 3, h = bh & 7;
  const size_t src = ((size_t)bh * Ln + l) * HDn + lane * 4;
  float4 ov = *(const float4*)(og + src);
  float ss = ov.x * ov.x + ov.y * ov.y + ov.z * ov.z + ov.w * ov.w;
  ss = wave_sum(ss);
  float sc = rsqrtf(ss * (1.f / 128.f) + EPSf);
  const float4 wv = *(const float4*)(w + lane * 4);
  const size_t dst = ((size_t)(b * Ln + l)) * Dn + h * HDn + lane * 4;
  on[dst + 0] = (bf16)(ov.x * sc * wv.x);
  on[dst + 1] = (bf16)(ov.y * sc * wv.y);
  on[dst + 2] = (bf16)(ov.z * sc * wv.z);
  on[dst + 3] = (bf16)(ov.w * sc * wv.w);
}

// ---------- host launcher ----------

extern "C" void kernel_launch(void* const* d_in, const int* in_sizes, int n_in,
                              void* d_out, int out_size, void* d_ws, size_t ws_size,
                              hipStream_t stream) {
  (void)in_sizes; (void)n_in; (void)out_size; (void)ws_size;
  const float* x     = (const float*)d_in[0];
  const float* Wq    = (const float*)d_in[1];
  const float* bq    = (const float*)d_in[2];
  const float* Wk    = (const float*)d_in[3];
  const float* bk    = (const float*)d_in[4];
  const float* Wv    = (const float*)d_in[5];
  const float* bv    = (const float*)d_in[6];
  const float* Wbeta = (const float*)d_in[7];
  const float* bbeta = (const float*)d_in[8];
  const float* Wo    = (const float*)d_in[9];
  const float* bo    = (const float*)d_in[10];
  const float* cq    = (const float*)d_in[11];
  const float* cbq   = (const float*)d_in[12];
  const float* ck    = (const float*)d_in[13];
  const float* cbk   = (const float*)d_in[14];
  const float* cv    = (const float*)d_in[15];
  const float* cbv   = (const float*)d_in[16];
  const float* rmsw  = (const float*)d_in[17];

  char* ws = (char*)d_ws;
  const size_t MB = 1ull << 20;
  // phase-1 buffers
  float* qp = (float*)(ws + 0 * MB);
  float* kp = (float*)(ws + 32 * MB);
  float* vp = (float*)(ws + 64 * MB);
  float* qc = (float*)(ws + 96 * MB);
  float* kc = (float*)(ws + 128 * MB);
  float* vc = (float*)(ws + 160 * MB);
  float* beta_raw = (float*)(ws + 192 * MB);
  // bf16-packed operands
  bf16* x_bf  = (bf16*)(ws + 193 * MB);   // 16MB
  bf16* Wq_bf = (bf16*)(ws + 209 * MB);   // 2MB each
  bf16* Wk_bf = (bf16*)(ws + 211 * MB);
  bf16* Wv_bf = (bf16*)(ws + 213 * MB);
  bf16* Wo_bf = (bf16*)(ws + 215 * MB);
  // phase-2 buffers (reuse consumed regions)
  bf16*  q_r    = (bf16*)(ws + 0 * MB);
  bf16*  k_r    = (bf16*)(ws + 16 * MB);
  float* v_r    = (float*)(ws + 32 * MB);
  float* beta_s = (float*)(ws + 64 * MB);
  float* og     = (float*)(ws + 96 * MB);
  bf16*  on_bf  = (bf16*)(ws + 128 * MB);

  const int M = Bn * Ln;        // 8192
  dim3 gg(Dn / 128, M / 128);   // (8, 64)

  // pack x and weights to bf16
  pack_bf16<<<(M * Dn) / 1024, 256, 0, stream>>>(x, x_bf);
  pack_bf16<<<(Dn * Dn) / 1024, 256, 0, stream>>>(Wq, Wq_bf);
  pack_bf16<<<(Dn * Dn) / 1024, 256, 0, stream>>>(Wk, Wk_bf);
  pack_bf16<<<(Dn * Dn) / 1024, 256, 0, stream>>>(Wv, Wv_bf);
  pack_bf16<<<(Dn * Dn) / 1024, 256, 0, stream>>>(Wo, Wo_bf);

  gemm_bf<<<gg, 256, 65536, stream>>>(x_bf, Wq_bf, bq, qp, M, Dn, Dn);
  gemm_bf<<<gg, 256, 65536, stream>>>(x_bf, Wk_bf, bk, kp, M, Dn, Dn);
  gemm_bf<<<gg, 256, 65536, stream>>>(x_bf, Wv_bf, bv, vp, M, Dn, Dn);
  beta_proj<<<M / 32, 256, 0, stream>>>(x, Wbeta, bbeta, beta_raw);

  const int convBlocks = (int)(((size_t)Bn * Ln * Dn) / 256);
  dwconv<<<convBlocks, 256, 0, stream>>>(qp, cq, cbq, qc);
  dwconv<<<convBlocks, 256, 0, stream>>>(kp, ck, cbk, kc);
  dwconv<<<convBlocks, 256, 0, stream>>>(vp, cv, cbv, vc);

  prep<<<(Bn * Hn * Ln) / 8, 256, 0, stream>>>(qc, kc, vc, beta_raw,
                                               q_r, k_r, v_r, beta_s);

  delta_scan<<<Bn * Hn, 256, SMEM_BYTES, stream>>>(q_r, k_r, v_r, beta_s, og);

  rmsnorm_k<<<(Bn * Hn * Ln) / 8, 256, 0, stream>>>(og, rmsw, on_bf);

  gemm_bf<<<gg, 256, 65536, stream>>>(on_bf, Wo_bf, bo, (float*)d_out, M, Dn, Dn);
}
